// DDPM_39831526703349
// MI455X (gfx1250) — matrix operand order repaired and checked by reading, not compiled, not run on hardware
//
#include <hip/hip_runtime.h>

typedef _Float16 f16;
typedef __attribute__((ext_vector_type(16))) _Float16 v16h;
typedef __attribute__((ext_vector_type(8)))  _Float16 v8h;
typedef __attribute__((ext_vector_type(8)))  float    v8f;

#define HWPIX 4096   // 64*64
#define NIMG  8
#define NPIX  (NIMG * HWPIX)  // 32768

// ---------------------------------------------------------------------------
// Input activation described as up to two channel-concatenated segments.
// Global channel ic < cUse -> segment A, else segment B at (ic - cUse).
// Each segment tensor is [N][cStride][64][64] f16.
// Segment sizes are multiples of 128, so a 32-wide ic tile never straddles.
// ---------------------------------------------------------------------------
struct Seg { const f16* p; int cUse; int cStride; };

// ---------------------------------------------------------------------------
// Implicit-GEMM conv via WMMA f32_16x16x32_f16.
//   D[oc, px] = sum_k Wt[oc,k] * Act[k, px],  K order = (kh, kw, ic)
// Weights pre-packed as [oc][kh][kw][ic] f16, so each 32-k tile is one
// spatial tap with 32 consecutive input channels.
// Block: 256 threads (8 waves). Tile: 64 OC x 64 pixels (= one image row).
// Wave w: mTile = w&3 (16 rows), two 16-pixel groups g = (w>>2)*2 + {0,1}.
// A (weights): double-buffered async global->LDS (ASYNCcnt), 2 in flight.
// B (act):     register-staged gather, tile t+1 overlaps tile t WMMA.
// ---------------------------------------------------------------------------
template <int KSZ>
__global__ __launch_bounds__(256) void wmma_conv(
    Seg sA, Seg sB, const f16* __restrict__ Wt,
    const float* __restrict__ scale, const float* __restrict__ bias, int relu,
    f16* out16, int oCoff, int oCtot, float* out32,
    int OC, int inC)
{
  __shared__ __align__(32) f16 As[2][64][32];    // weights: [buf][m][k]
  __shared__ __align__(32) f16 Bs[4][16][32];    // act:     [grp][n][k]

  const int t    = threadIdx.x;
  const int wv   = t >> 5;
  const int lane = t & 31;
  const int mTile = wv & 3;
  const int gBase = (wv >> 2) * 2;

  const int ocBase = blockIdx.y * 64;
  const int pxBase = blockIdx.x * 64;   // one full image row
  const int pn = pxBase >> 12;          // image index (block-uniform)
  const int ph = (pxBase >> 6) & 63;    // row index   (block-uniform)

  // loader assignment: thread -> 8 contiguous k (=ic) for one row/pixel
  const int am  = t >> 2;               // 0..63 weight row
  const int ak0 = (t & 3) * 8;          // 0,8,16,24
  const int bpx = t >> 2;               // 0..63 pixel within the row (== w)
  const int bk0 = (t & 3) * 8;          // ic sub-offset

  const int PAD = (KSZ == 3) ? 1 : 0;
  const int K   = inC * KSZ * KSZ;
  const int nT  = K >> 5;               // 32-k tiles
  const long wrow = (long)(ocBase + am) * K;

  v8f acc0 = {}; v8f acc1 = {};

  // ---- async weight-tile copy: global -> LDS, tracked by ASYNCcnt ----
  auto asyncA = [&](int tt, int buf) {
    unsigned lds = (unsigned)(unsigned long long)(&As[buf][am][ak0]);
    unsigned vo  = (unsigned)((wrow + (long)tt * 32 + ak0) * 2);
    asm volatile("global_load_async_to_lds_b128 %0, %1, %2"
                 :: "v"(lds), "v"(vo), "s"(Wt) : "memory");
  };

  // ---- register-staged activation tile (sp = spatial tap, icb = ic base) --
  auto loadB = [&](int sp, int icb, v8h& bb) {
    const int kh = sp / KSZ, kw = sp - kh * KSZ;
    const int hh = ph + kh - PAD;
    const int ww = bpx + kw - PAD;
    const bool valid = ((unsigned)hh < 64u) & ((unsigned)ww < 64u);
    const int hc = hh < 0 ? 0 : (hh > 63 ? 63 : hh);
    const int wc = ww < 0 ? 0 : (ww > 63 ? 63 : ww);

    int ic0 = icb + bk0;
    const f16* p; int cs;
    if (ic0 < sA.cUse) { p = sA.p; cs = sA.cStride; }
    else               { p = sB.p; cs = sB.cStride; ic0 -= sA.cUse; }
    const f16* base = p + ((long)pn * cs + ic0) * HWPIX + hc * 64 + wc;
#pragma unroll
    for (int i = 0; i < 8; ++i) {
      f16 v = base[(long)i * HWPIX];              // clamped-safe address
      bb[i] = valid ? v : (f16)0.f;
    }
  };

  // ---- prologue: tile 0 in flight ----
  asyncA(0, 0);
  v8h bb;
  int sp = 0, icb = 0;
  loadB(sp, icb, bb);

  for (int tt = 0; tt < nT; ++tt) {
    const int buf = tt & 1;

    // stage current B tile into LDS
    *(v8h*)&Bs[bpx >> 4][bpx & 15][bk0] = bb;

    // issue next A tile (other buffer), then drain so tile tt has landed
    // while tile tt+1 stays in flight (async loads complete in order)
    if (tt + 1 < nT) {
      asyncA(tt + 1, buf ^ 1);
      asm volatile("s_wait_asynccnt 0x1" ::: "memory");
    } else {
      asm volatile("s_wait_asynccnt 0x0" ::: "memory");
    }
    __syncthreads();

    // prefetch next B tile into registers (overlaps WMMA below)
    if (tt + 1 < nT) {
      icb += 32;
      if (icb == inC) { icb = 0; ++sp; }
      loadB(sp, icb, bb);
    }

    // ---- fragments (contiguous k = 16*(lane>>4) + s, same order in A and B) ----
    const int m  = mTile * 16 + (lane & 15);
    const int kg = lane >> 4;
    v16h a = *(const v16h*)&As[buf][m][kg * 16];

    v16h b0 = *(const v16h*)&Bs[gBase + 0][lane & 15][kg * 16];
    v16h b1 = *(const v16h*)&Bs[gBase + 1][lane & 15][kg * 16];

    acc0 = __builtin_amdgcn_wmma_f32_16x16x32_f16(false, a, false, b0,
                                                  (short)0, acc0, false, false);
    acc1 = __builtin_amdgcn_wmma_f32_16x16x32_f16(false, a, false, b1,
                                                  (short)0, acc1, false, false);
    __syncthreads();
  }

  // ---- epilogue: scale/bias/relu, write f16 (chain) and/or f32 (final) ----
  const int n0 = lane & 15;
  const int mo = 8 * (lane >> 4);
#pragma unroll
  for (int td = 0; td < 2; ++td) {
    v8f acc = td ? acc1 : acc0;
    const int w2 = (gBase + td) * 16 + n0;       // w coordinate in the row
#pragma unroll
    for (int r = 0; r < 8; ++r) {
      int oc = ocBase + mTile * 16 + mo + r;
      float v = acc[r];
      if (scale) v *= scale[oc];
      if (bias)  v += bias[oc];
      if (relu)  v = v > 0.f ? v : 0.f;
      long oidx = ((long)pn * oCtot + oCoff + oc) * HWPIX + ph * 64 + w2;
      if (out16) out16[oidx] = (f16)v;
      if (out32) out32[oidx] = v;
    }
  }
}

// ---------------------------------------------------------------------------
// f32 -> f16 conversion (input activations)
// ---------------------------------------------------------------------------
__global__ void cvt_f32_f16(const float* __restrict__ in, f16* __restrict__ out, long n) {
  long i = (long)blockIdx.x * blockDim.x + threadIdx.x;
  long s = (long)gridDim.x * blockDim.x;
  for (; i < n; i += s) out[i] = (f16)in[i];
}

// ---------------------------------------------------------------------------
// Weight pack: [oc][ic][kh][kw] f32 -> [oc][kh][kw][ic] f16
// ---------------------------------------------------------------------------
__global__ void pack_w(const float* __restrict__ w, f16* __restrict__ out,
                       int inC, int k9, long total) {
  long j = (long)blockIdx.x * blockDim.x + threadIdx.x;
  long s = (long)gridDim.x * blockDim.x;
  long krow = (long)inC * k9;
  for (; j < total; j += s) {
    long oc = j / krow;
    long r  = j - oc * krow;
    int  sp = (int)(r / inC);
    int  ic = (int)(r - (long)sp * inC);
    out[j] = (f16)w[(oc * inC + ic) * k9 + sp];
  }
}

// ---------------------------------------------------------------------------
// Fold BN (inference) into per-channel scale/bias.
// ---------------------------------------------------------------------------
__global__ void bn_prep(const float* __restrict__ g, const float* __restrict__ b,
                        const float* __restrict__ m, const float* __restrict__ v,
                        float* __restrict__ scale, float* __restrict__ bias, int C) {
  int i = blockIdx.x * blockDim.x + threadIdx.x;
  if (i < C) {
    float s = g[i] * rsqrtf(v[i] + 1e-5f);
    scale[i] = s;
    bias[i]  = b[i] - m[i] * s;
  }
}

// ---------------------------------------------------------------------------
// Per-pixel dynamic depthwise 3x3 conv, dilation d.
// xcat: [N][512][64][64] f16 (xd lives in channels 0..127)
// ker : [N][1152][64][64] f16 (C*9 kernel planes)
// res : [N][128][64][64] f16
// ---------------------------------------------------------------------------
__global__ __launch_bounds__(256) void dyn_conv_k(
    const f16* __restrict__ xcat, const f16* __restrict__ ker,
    f16* __restrict__ res, int d)
{
  int idx = blockIdx.x * 256 + threadIdx.x;   // N*128*4096 total
  int w = idx & 63, h = (idx >> 6) & 63;
  int c = (idx >> 12) & 127, n = idx >> 19;
  float acc = 0.f;
#pragma unroll
  for (int ki = 0; ki < 3; ++ki)
#pragma unroll
    for (int kj = 0; kj < 3; ++kj) {
      int hh = h + (ki - 1) * d, ww = w + (kj - 1) * d;
      if ((unsigned)hh < 64u && (unsigned)ww < 64u) {
        float xv = (float)xcat[(((long)n * 512 + c) * HWPIX) + hh * 64 + ww];
        float kv = (float)ker[(((long)n * 1152 + c * 9 + ki * 3 + kj) * HWPIX) + h * 64 + w];
        acc += xv * kv;
      }
    }
  res[idx] = (f16)acc;
}

// ---------------------------------------------------------------------------
// Host driver
// ---------------------------------------------------------------------------
extern "C" void kernel_launch(void* const* d_in, const int* in_sizes, int n_in,
                              void* d_out, int out_size, void* d_ws, size_t ws_size,
                              hipStream_t stream) {
  (void)in_sizes; (void)n_in; (void)out_size; (void)ws_size;

  auto F = [&](int i) { return (const float*)d_in[i]; };

  const size_t MB = 1024ull * 1024ull;
  char* ws = (char*)d_ws;
  f16*  y16   = (f16*)(ws + 0);          // 32 MB  [8][512][4096]
  f16*  cat16 = (f16*)(ws + 32  * MB);   // 32 MB  [8][512][4096]  xd|r1|r3|r5
  f16*  g0o   = (f16*)(ws + 64  * MB);   // 32 MB  [8][512][4096]
  f16*  feats = (f16*)(ws + 96  * MB);   // 24 MB  [8][384][4096]  f1|f2|f3
  f16*  dn16  = (f16*)(ws + 120 * MB);   // 8  MB  [8][128][4096]
  f16*  f4    = (f16*)(ws + 128 * MB);   // 8  MB  [8][128][4096]
  f16*  dfo   = (f16*)(ws + 136 * MB);   // 32 MB  [8][512][4096]
  f16*  g2o   = (f16*)(ws + 168 * MB);   // 72 MB  [8][1152][4096] (x16 early)
  f16*  res16 = (f16*)(ws + 240 * MB);   // 8  MB  [8][128][4096]
  f16*  wpk   = (f16*)(ws + 248 * MB);   // 8  MB  packed weights
  float* sbS  = (float*)(ws + 256 * MB); // scale
  float* sbB  = sbS + 2048;              // bias

  auto gemm = [&](Seg A, Seg B, const float* w32, int OC, int inC, int ksz,
                  const float* scl, const float* bs, int relu,
                  f16* o16, int oCoff, int oCtot, float* o32) {
    long nW = (long)OC * inC * ksz * ksz;
    pack_w<<<2048, 256, 0, stream>>>(w32, wpk, inC, ksz * ksz, nW);
    dim3 grid(NPIX / 64, OC / 64);
    if (ksz == 3)
      wmma_conv<3><<<grid, 256, 0, stream>>>(A, B, wpk, scl, bs, relu,
                                             o16, oCoff, oCtot, o32, OC, inC);
    else
      wmma_conv<1><<<grid, 256, 0, stream>>>(A, B, wpk, scl, bs, relu,
                                             o16, oCoff, oCtot, o32, OC, inC);
  };
  auto bnsb = [&](int gi, int C) {  // gi = index of gamma; beta/mean/var follow
    bn_prep<<<(C + 255) / 256, 256, 0, stream>>>(F(gi), F(gi + 1), F(gi + 2),
                                                 F(gi + 3), sbS, sbB, C);
  };

  Seg none{nullptr, 0, 0};
  const long actN = 8L * 512 * HWPIX;

  // ---- input conversions ----
  cvt_f32_f16<<<4096, 256, 0, stream>>>(F(0), g2o, actN);   // x -> f16 (temp in g2o)
  cvt_f32_f16<<<4096, 256, 0, stream>>>(F(1), y16, actN);   // y -> f16

  // ---- xd = 1x1 down (512->128), write into cat16 channels [0,128) ----
  gemm({g2o, 512, 512}, none, F(2), 128, 512, 1, nullptr, F(3), 0,
       cat16, 0, 512, nullptr);

  // ---- branches ----
  // flattened param index layout per branch (base b):
  //   b+0 g0_w, b+1 g0_b, b+2 dd_w, b+3 dd_b,
  //   b+4+i*5 block_i.w (gamma/beta/mean/var follow),
  //   b+24 dfuse.w (+bn), b+29 g2_w, b+30 g2_b, b+31 fuse_w, b+32 fuse_b
  auto run_branch = [&](int b, int d, int slot) {
    // g0: 3x3 512->512 (+bias)
    gemm({y16, 512, 512}, none, F(b + 0), 512, 512, 3, nullptr, F(b + 1), 0,
         g0o, 0, 512, nullptr);
    // dense down: 1x1 512->128 (+bias)
    gemm({g0o, 512, 512}, none, F(b + 2), 128, 512, 1, nullptr, F(b + 3), 0,
         dn16, 0, 128, nullptr);
    // dense blocks: 3x3, in = [f1..fi | down], BN+ReLU
    for (int i = 0; i < 4; ++i) {
      int wb = b + 4 + i * 5;
      bnsb(wb + 1, 128);
      f16* o  = (i < 3) ? feats : f4;
      int off = (i < 3) ? i * 128 : 0;
      int ct  = (i < 3) ? 384 : 128;
      gemm({feats, i * 128, 384}, {dn16, 128, 128}, F(wb),
           128, (i + 1) * 128, 3, sbS, sbB, 1, o, off, ct, nullptr);
    }
    // dense fuse: 3x3, in = [g0 | f4] (640) -> 512, BN+ReLU
    bnsb(b + 25, 512);
    gemm({g0o, 512, 512}, {f4, 128, 128}, F(b + 24), 512, 640, 3,
         sbS, sbB, 1, dfo, 0, 512, nullptr);
    // g2: 1x1 512->1152 (+bias)
    gemm({dfo, 512, 512}, none, F(b + 29), 1152, 512, 1, nullptr, F(b + 30), 0,
         g2o, 0, 1152, nullptr);
    // dynamic depthwise conv on xd (cat16 channels 0..127)
    dyn_conv_k<<<(8 * 128 * HWPIX) / 256, 256, 0, stream>>>(cat16, g2o, res16, d);
    // branch fuse: 3x3 128->128 (+bias) -> cat16 slice [slot*128, slot*128+128)
    gemm({res16, 128, 128}, none, F(b + 31), 128, 128, 3, nullptr, F(b + 32), 0,
         cat16, slot * 128, 512, nullptr);
  };

  run_branch(4, 1, 1);
  run_branch(37, 3, 2);
  run_branch(70, 5, 3);

  // ---- final fuse: 3x3 512->512, BN+ReLU, f32 output ----
  bnsb(104, 512);
  gemm({cat16, 512, 512}, none, F(103), 512, 512, 3, sbS, sbB, 1,
       nullptr, 0, 512, (float*)d_out);
}
